// MultiHeadSelfAttention_40776419508319
// MI455X (gfx1250) — matrix operand order repaired and checked by reading, not compiled, not run on hardware
//
#include <hip/hip_runtime.h>
#include <math.h>

typedef __bf16 bf16_t;
typedef bf16_t bf16x16 __attribute__((ext_vector_type(16)));
typedef float  floatx8 __attribute__((ext_vector_type(8)));
typedef unsigned uint32x4 __attribute__((ext_vector_type(4)));
typedef int int32x8 __attribute__((ext_vector_type(8)));
typedef int int32x4 __attribute__((ext_vector_type(4)));

#define USE_TDM 1

// ---------- helpers ----------

static __device__ __forceinline__ bf16_t f2bf(float f) {
    union { float f; unsigned u; } in; in.f = f;
    unsigned u = in.u;
    unsigned r = (u + 0x7FFFu + ((u >> 16) & 1u)) >> 16;
    union { unsigned short s; bf16_t b; } out;
    out.s = (unsigned short)r;
    return out.b;
}

// Assemble a 16x bf16 WMMA fragment from two 16-byte chunks.
static __device__ __forceinline__ bf16x16 ld_frag(const bf16_t* p0, const bf16_t* p1) {
    union { uint4 u[2]; bf16x16 v; } r;
    r.u[0] = *(const uint4*)p0;
    r.u[1] = *(const uint4*)p1;
    return r.v;
}

// DPP ROW_XMASK butterfly (xor within 16-lane rows) -- pure-VALU reductions.
template <int CTRL>
static __device__ __forceinline__ float dpp_xor_f(float x) {
    int i = __float_as_int(x);
    i = __builtin_amdgcn_update_dpp(0, i, CTRL, 0xf, 0xf, true);
    return __int_as_float(i);
}
static __device__ __forceinline__ float red_max16(float x) {
    x = fmaxf(x, dpp_xor_f<0x161>(x));   // row_xmask:1
    x = fmaxf(x, dpp_xor_f<0x162>(x));   // row_xmask:2
    x = fmaxf(x, dpp_xor_f<0x164>(x));   // row_xmask:4
    x = fmaxf(x, dpp_xor_f<0x168>(x));   // row_xmask:8
    return x;
}
static __device__ __forceinline__ float red_sum16(float x) {
    x += dpp_xor_f<0x161>(x);
    x += dpp_xor_f<0x162>(x);
    x += dpp_xor_f<0x164>(x);
    x += dpp_xor_f<0x168>(x);
    return x;
}

#if USE_TDM
// Tensor Data Mover: async 2D bf16 tile load Global -> LDS with HW padding.
// pad_interval=32 dwords (one 64-elem row), pad_amount=4 dwords (8 elems)
// reproduces the 72-element padded LDS stride.
static __device__ __forceinline__ unsigned lds_off(const void* p) {
    return (unsigned)(size_t)p;     // generic LDS pointer: low 32 bits = LDS offset
}
static __device__ __forceinline__ void tdm_load_2d(
    unsigned lds_byte_off, const void* gptr, unsigned ld_elems,
    unsigned rows, unsigned cols)
{
    unsigned long long ga = (unsigned long long)(size_t)gptr;
    uint32x4 g0;
    g0.x = 1u;                                   // count=1, user descriptor
    g0.y = lds_byte_off;                         // lds_addr
    g0.z = (unsigned)ga;                         // global_addr[31:0]
    g0.w = (unsigned)(ga >> 32) | (2u << 30);    // global_addr[56:32] | type=2
    int32x8 g1;
    g1[0] = (int)((1u << 16)                     // data_size = 2 bytes
                | (1u << 20)                     // pad_enable
                | (4u << 22)                     // pad_interval = 32 dwords
                | (3u << 25));                   // pad_amount  = 4 dwords
    g1[1] = 0;                                   // tensor_dim0 = 1<<30 (low16=0)
    g1[2] = (int)0x4000u;                        // tensor_dim0 hi | tensor_dim1 lo
    g1[3] = (int)(0x4000u | (cols << 16));       // tensor_dim1 hi | tile_dim0
    g1[4] = (int)rows;                           // tile_dim1 (tile_dim2 = 0)
    g1[5] = (int)ld_elems;                       // tensor_dim0_stride low 32
    g1[6] = 0;
    g1[7] = 0;
    int32x4 z4; z4[0] = z4[1] = z4[2] = z4[3] = 0;
    int32x8 z8;
    #pragma unroll
    for (int i = 0; i < 8; ++i) z8[i] = 0;
    __builtin_amdgcn_tensor_load_to_lds(g0, g1, z4, z4, z8, 0);
}
#endif

// ---------- weight convert + transpose (f32 [K x N] -> bf16 [N x K]) ----------

__global__ __launch_bounds__(256) void transpose_cast_kernel(
    const float* __restrict__ in, bf16_t* __restrict__ out, int K, int N)
{
    __shared__ float tile[32][33];
    const int bx = blockIdx.x;            // over N
    const int by = blockIdx.y;            // over K
    const int x = threadIdx.x;            // 0..31
    const int y = threadIdx.y;            // 0..7
    #pragma unroll
    for (int j = 0; j < 4; ++j) {
        int r = by * 32 + y + j * 8;
        tile[y + j * 8][x] = in[(size_t)r * N + bx * 32 + x];
    }
    __syncthreads();
    #pragma unroll
    for (int j = 0; j < 4; ++j) {
        int n = bx * 32 + y + j * 8;
        out[(size_t)n * K + by * 32 + x] = f2bf(tile[x][y + j * 8]);
    }
}

// ---------- LayerNorm (block per row), optional f32 / bf16 outputs ----------

__global__ __launch_bounds__(256) void ln_kernel(
    const float* __restrict__ in, const float* __restrict__ g,
    const float* __restrict__ bta, float* __restrict__ outF,
    bf16_t* __restrict__ outB, int C)
{
    const int row = blockIdx.x;
    const int tid = threadIdx.x;
    const float4 v = ((const float4*)(in + (size_t)row * C))[tid];
    float s  = v.x + v.y + v.z + v.w;
    float s2 = v.x * v.x + v.y * v.y + v.z * v.z + v.w * v.w;
    // in-wave butterfly: xor 1..8 via DPP, xor16 via shfl
    s  = red_sum16(s);
    s2 = red_sum16(s2);
    s  += __shfl_xor(s, 16, 32);
    s2 += __shfl_xor(s2, 16, 32);
    __shared__ float red0[8], red1[8];
    const int wave = tid >> 5, lane = tid & 31;
    if (lane == 0) { red0[wave] = s; red1[wave] = s2; }
    __syncthreads();
    if (wave == 0) {
        float a = (lane < 8) ? red0[lane] : 0.0f;
        float b = (lane < 8) ? red1[lane] : 0.0f;
        #pragma unroll
        for (int m = 1; m < 8; m <<= 1) {
            a += __shfl_xor(a, m, 32);
            b += __shfl_xor(b, m, 32);
        }
        if (lane == 0) { red0[0] = a; red1[0] = b; }
    }
    __syncthreads();
    const float mu  = red0[0] / (float)C;
    const float var = red1[0] / (float)C - mu * mu;
    const float rs  = rsqrtf(var + 1e-5f);
    const float4 gv = ((const float4*)g)[tid];
    const float4 bv = ((const float4*)bta)[tid];
    float4 o;
    o.x = (v.x - mu) * rs * gv.x + bv.x;
    o.y = (v.y - mu) * rs * gv.y + bv.y;
    o.z = (v.z - mu) * rs * gv.z + bv.z;
    o.w = (v.w - mu) * rs * gv.w + bv.w;
    if (outF) ((float4*)(outF + (size_t)row * C))[tid] = o;
    if (outB) {
        bf16_t* ob = outB + (size_t)row * C + tid * 4;
        ob[0] = f2bf(o.x); ob[1] = f2bf(o.y); ob[2] = f2bf(o.z); ob[3] = f2bf(o.w);
    }
}

// ---------- bf16 WMMA GEMM: out = A[MxK] * Bt[NxK]^T + bias (+res) (gelu?) ----------
// 128 threads = 4 waves; block tile 128x64; wave tile 32x64; BK=64;
// double-buffered LDS staged by the Tensor Data Mover (TENSORcnt-tracked).

#define GBM 128
#define GBN 64
#define GBK 64
#define AST 72
#define BST 72

template <bool GELU>
__global__ __launch_bounds__(128) void gemm_bf16_kernel(
    const bf16_t* __restrict__ A, const bf16_t* __restrict__ Bt,
    const float* __restrict__ bias, const float* __restrict__ residual,
    float* __restrict__ outF, bf16_t* __restrict__ outB,
    int M, int N, int K)
{
    __shared__ bf16_t As[2][GBM * AST];
    __shared__ bf16_t Bs[2][GBN * BST];

    const int tid  = threadIdx.x;
    const int lane = tid & 31;
    const int wave = tid >> 5;       // 0..3: 32 rows each
    const int lo   = lane & 15;
    const int hi   = lane >> 4;
    const int m0   = blockIdx.y * GBM;
    const int n0   = blockIdx.x * GBN;

    floatx8 acc[2][4];
    #pragma unroll
    for (int i = 0; i < 2; ++i)
        #pragma unroll
        for (int j = 0; j < 4; ++j)
            #pragma unroll
            for (int e = 0; e < 8; ++e) acc[i][j][e] = 0.0f;

    const int nk = K / GBK;
#if USE_TDM
    if (wave == 0) {
        tdm_load_2d(lds_off(&As[0][0]), A + (size_t)m0 * K, K, GBM, GBK);
        tdm_load_2d(lds_off(&Bs[0][0]), Bt + (size_t)n0 * K, K, GBN, GBK);
    }
#endif
    for (int ki = 0; ki < nk; ++ki) {
        const int buf = ki & 1;
        const int k0 = ki * GBK;
#if USE_TDM
        if (wave == 0) {
            if (ki + 1 < nk) {
                tdm_load_2d(lds_off(&As[buf ^ 1][0]),
                            A + (size_t)m0 * K + k0 + GBK, K, GBM, GBK);
                tdm_load_2d(lds_off(&Bs[buf ^ 1][0]),
                            Bt + (size_t)n0 * K + k0 + GBK, K, GBN, GBK);
                __builtin_amdgcn_s_wait_tensorcnt(2);   // current pair complete
            } else {
                __builtin_amdgcn_s_wait_tensorcnt(0);
            }
        }
        __syncthreads();
#else
        __syncthreads();
        #pragma unroll
        for (int it = 0; it < (GBM * GBK) / (128 * 8); ++it) {
            int idx = (tid + it * 128) * 8;
            int r = idx / GBK, c = idx % GBK;
            uint4 d = *(const uint4*)(A + (size_t)(m0 + r) * K + k0 + c);
            *(uint4*)(&As[buf][0] + r * AST + c) = d;
        }
        #pragma unroll
        for (int it = 0; it < (GBN * GBK) / (128 * 8); ++it) {
            int idx = (tid + it * 128) * 8;
            int r = idx / GBK, c = idx % GBK;
            uint4 d = *(const uint4*)(Bt + (size_t)(n0 + r) * K + k0 + c);
            *(uint4*)(&Bs[buf][0] + r * BST + c) = d;
        }
        __syncthreads();
#endif
        #pragma unroll
        for (int ks = 0; ks < GBK / 32; ++ks) {
            bf16x16 af[2], bfg[4];
            #pragma unroll
            for (int i = 0; i < 2; ++i) {
                const bf16_t* pr = &As[buf][0] + (wave * 32 + i * 16 + lo) * AST + ks * 32;
                af[i] = ld_frag(pr + hi * 8, pr + 16 + hi * 8);
            }
            #pragma unroll
            for (int j = 0; j < 4; ++j) {
                const bf16_t* pr = &Bs[buf][0] + (j * 16 + lo) * BST + ks * 32;
                bfg[j] = ld_frag(pr + hi * 8, pr + 16 + hi * 8);
            }
            #pragma unroll
            for (int i = 0; i < 2; ++i)
                #pragma unroll
                for (int j = 0; j < 4; ++j)
                    acc[i][j] = __builtin_amdgcn_wmma_f32_16x16x32_bf16(
                        false, af[i], false, bfg[j], (short)0, acc[i][j],
                        false, false);
        }
        __syncthreads();   // all waves done before next tile overwrites LDS
    }

    #pragma unroll
    for (int i = 0; i < 2; ++i) {
        #pragma unroll
        for (int j = 0; j < 4; ++j) {
            const int colg = n0 + j * 16 + lo;
            const float bz = bias ? bias[colg] : 0.0f;
            #pragma unroll
            for (int r = 0; r < 8; ++r) {
                const int rowg = m0 + wave * 32 + i * 16 + r + 8 * hi;
                const size_t off = (size_t)rowg * N + colg;
                float v = acc[i][j][r] + bz;
                if (residual) v += residual[off];
                if (GELU) v = 0.5f * v * (1.0f + erff(v * 0.70710678118654752f));
                if (outF) outF[off] = v;
                if (outB) outB[off] = f2bf(v);
            }
        }
    }
}

// ---------- flash attention (bf16 WMMA, f32 online softmax, DPP reductions) ----------

#define KST 72
#define VST 40
#define PST 40

__global__ __launch_bounds__(256) void attn_kernel(
    const bf16_t* __restrict__ Q, const bf16_t* __restrict__ Kb,
    const bf16_t* __restrict__ V, bf16_t* __restrict__ O,
    int Nseq, int C, int H, float scale)
{
    __shared__ bf16_t Ks[32 * KST];       // K chunk  [kv=32][d=64]
    __shared__ bf16_t Vt[64 * VST];       // V^T chunk [d=64][kv=32]
    __shared__ bf16_t Ps[8 * 16 * PST];   // per-wave P tile [16][32]

    const int tid  = threadIdx.x;
    const int lane = tid & 31, wave = tid >> 5;
    const int lo   = lane & 15, hi = lane >> 4;
    const int bh   = blockIdx.x;
    const int b    = bh / H, h = bh % H;
    const int q0   = blockIdx.y * 128 + wave * 16;   // query tile base
    const size_t rowbase = (size_t)b * Nseq;
    const int c0   = h * 64;

    // Q fragments: 16 rows x 64 (two K=32 steps)
    bf16x16 qa[2];
    {
        const bf16_t* qp = Q + (rowbase + q0 + lo) * C + c0;
        #pragma unroll
        for (int ks = 0; ks < 2; ++ks)
            qa[ks] = ld_frag(qp + ks * 32 + hi * 8, qp + ks * 32 + 16 + hi * 8);
    }

    floatx8 oacc[4];
    #pragma unroll
    for (int t = 0; t < 4; ++t)
        #pragma unroll
        for (int e = 0; e < 8; ++e) oacc[t][e] = 0.0f;
    float mrun[8], lrun[8];
    #pragma unroll
    for (int r = 0; r < 8; ++r) { mrun[r] = -1e30f; lrun[r] = 0.0f; }

    bf16_t* Pw = Ps + wave * 16 * PST;

    for (int kv0 = 0; kv0 < Nseq; kv0 += 32) {
        __syncthreads();
        {   // cooperative stage of K chunk and V^T chunk (8 bf16 per thread each)
            const int idx = tid * 8;
            const int kv = idx / 64, d0 = idx % 64;
            uint4 kd = *(const uint4*)(Kb + (rowbase + kv0 + kv) * C + c0 + d0);
            *(uint4*)(Ks + kv * KST + d0) = kd;
            uint4 vd = *(const uint4*)(V + (rowbase + kv0 + kv) * C + c0 + d0);
            const bf16_t* ve = (const bf16_t*)&vd;
            #pragma unroll
            for (int e = 0; e < 8; ++e) Vt[(d0 + e) * VST + kv] = ve[e];
        }
        __syncthreads();

        // S = Q * K^T : two 16x16 kv tiles, K-dim = 64 in two steps
        floatx8 s0, s1;
        #pragma unroll
        for (int e = 0; e < 8; ++e) { s0[e] = 0.0f; s1[e] = 0.0f; }
        #pragma unroll
        for (int ks = 0; ks < 2; ++ks) {
            const bf16_t* kp0 = Ks + (lo) * KST + ks * 32;
            const bf16_t* kp1 = Ks + (16 + lo) * KST + ks * 32;
            bf16x16 kb0 = ld_frag(kp0 + hi * 8, kp0 + 16 + hi * 8);
            bf16x16 kb1 = ld_frag(kp1 + hi * 8, kp1 + 16 + hi * 8);
            s0 = __builtin_amdgcn_wmma_f32_16x16x32_bf16(false, qa[ks], false, kb0,
                                                         (short)0, s0, false, false);
            s1 = __builtin_amdgcn_wmma_f32_16x16x32_bf16(false, qa[ks], false, kb1,
                                                         (short)0, s1, false, false);
        }

        // online softmax; rows r -> matrix row (r + 8*hi); DPP butterflies
        #pragma unroll
        for (int r = 0; r < 8; ++r) {
            float a = s0[r] * scale, bb = s1[r] * scale;
            float mx = red_max16(fmaxf(a, bb));
            float mnew  = fmaxf(mrun[r], mx);
            float alpha = __expf(mrun[r] - mnew);
            float p0 = __expf(a - mnew), p1 = __expf(bb - mnew);
            float ps = red_sum16(p0 + p1);
            lrun[r] = lrun[r] * alpha + ps;
            mrun[r] = mnew;
            #pragma unroll
            for (int t = 0; t < 4; ++t) oacc[t][r] = oacc[t][r] * alpha;
            const int mrow = r + 8 * hi;
            Pw[mrow * PST + lo]      = f2bf(p0);
            Pw[mrow * PST + 16 + lo] = f2bf(p1);
        }

        // O += P * V (K-dim = 32 kv), 4 d-tiles of 16
        {
            const bf16_t* pp = Pw + lo * PST;
            bf16x16 pfrag = ld_frag(pp + hi * 8, pp + 16 + hi * 8);
            #pragma unroll
            for (int t = 0; t < 4; ++t) {
                const bf16_t* vp = Vt + (t * 16 + lo) * VST;
                bf16x16 vb = ld_frag(vp + hi * 8, vp + 16 + hi * 8);
                oacc[t] = __builtin_amdgcn_wmma_f32_16x16x32_bf16(
                    false, pfrag, false, vb, (short)0, oacc[t], false, false);
            }
        }
    }

    // normalize and store O (bf16)
    #pragma unroll
    for (int r = 0; r < 8; ++r) {
        const float inv = 1.0f / lrun[r];
        const int mrow = r + 8 * hi;
        bf16_t* op = O + (rowbase + q0 + mrow) * C + c0;
        #pragma unroll
        for (int t = 0; t < 4; ++t)
            op[t * 16 + lo] = f2bf(oacc[t][r] * inv);
    }
}

// ---------- launch ----------

extern "C" void kernel_launch(void* const* d_in, const int* in_sizes, int n_in,
                              void* d_out, int out_size, void* d_ws, size_t ws_size,
                              hipStream_t stream)
{
    (void)in_sizes; (void)n_in; (void)out_size; (void)ws_size;
    const float* x    = (const float*)d_in[0];
    const float* wq   = (const float*)d_in[1];
    const float* bq   = (const float*)d_in[2];
    const float* wk   = (const float*)d_in[3];
    const float* bk   = (const float*)d_in[4];
    const float* wv   = (const float*)d_in[5];
    const float* bv   = (const float*)d_in[6];
    const float* wo   = (const float*)d_in[7];
    const float* bo   = (const float*)d_in[8];
    const float* ln1w = (const float*)d_in[9];
    const float* ln1b = (const float*)d_in[10];
    const float* ln2w = (const float*)d_in[11];
    const float* ln2b = (const float*)d_in[12];
    const float* fc1w = (const float*)d_in[13];
    const float* fc1b = (const float*)d_in[14];
    const float* fc2w = (const float*)d_in[15];
    const float* fc2b = (const float*)d_in[16];
    const float* wout = (const float*)d_in[17];
    const float* bout = (const float*)d_in[18];
    const float* lnow = (const float*)d_in[19];
    const float* lnob = (const float*)d_in[20];

    constexpr int B = 2, N = 2048, C = 1024, H = 16;
    constexpr int M = B * N, C4 = 4 * C;
    const size_t MB = (size_t)1 << 20;

    char* w = (char*)d_ws;
    bf16_t* wqT   = (bf16_t*)(w + 0 * MB);
    bf16_t* wkT   = (bf16_t*)(w + 2 * MB);
    bf16_t* wvT   = (bf16_t*)(w + 4 * MB);
    bf16_t* woT   = (bf16_t*)(w + 6 * MB);
    bf16_t* fc1T  = (bf16_t*)(w + 8 * MB);    // 8 MB
    bf16_t* fc2T  = (bf16_t*)(w + 16 * MB);   // 8 MB
    bf16_t* woutT = (bf16_t*)(w + 24 * MB);   // 2 MB
    bf16_t* hbf   = (bf16_t*)(w + 26 * MB);   // 8 MB (dead after QKV)
    bf16_t* qbf   = (bf16_t*)(w + 34 * MB);   // 8 MB
    bf16_t* kbf   = (bf16_t*)(w + 42 * MB);   // 8 MB
    bf16_t* vbf   = (bf16_t*)(w + 50 * MB);   // 8 MB (q/k/v dead after attn)
    bf16_t* obf   = (bf16_t*)(w + 58 * MB);   // 8 MB (dead after Wo GEMM)
    float*  h2    = (float*)(w + 66 * MB);    // 16 MB (dead after fc2)
    bf16_t* mbf   = (bf16_t*)(w + 82 * MB);   // 8 MB
    bf16_t* gbf   = (bf16_t*)(w + 26 * MB);   // 32 MB, reuses hbf+q+k+v
    bf16_t* h3bf  = (bf16_t*)(w + 58 * MB);   // reuses obf
    float*  pf    = (float*)(w + 66 * MB);    // reuses h2

    const dim3 tb(32, 8);
    transpose_cast_kernel<<<dim3(C / 32, C / 32), tb, 0, stream>>>(wq, wqT, C, C);
    transpose_cast_kernel<<<dim3(C / 32, C / 32), tb, 0, stream>>>(wk, wkT, C, C);
    transpose_cast_kernel<<<dim3(C / 32, C / 32), tb, 0, stream>>>(wv, wvT, C, C);
    transpose_cast_kernel<<<dim3(C / 32, C / 32), tb, 0, stream>>>(wo, woT, C, C);
    transpose_cast_kernel<<<dim3(C4 / 32, C / 32), tb, 0, stream>>>(fc1w, fc1T, C, C4);
    transpose_cast_kernel<<<dim3(C / 32, C4 / 32), tb, 0, stream>>>(fc2w, fc2T, C4, C);
    transpose_cast_kernel<<<dim3(C / 32, C / 32), tb, 0, stream>>>(wout, woutT, C, C);

    // ln1 -> bf16 activations
    ln_kernel<<<M, 256, 0, stream>>>(x, ln1w, ln1b, nullptr, hbf, C);

    // QKV projections
    gemm_bf16_kernel<false><<<dim3(C / GBN, M / GBM), 128, 0, stream>>>(
        hbf, wqT, bq, nullptr, nullptr, qbf, M, C, C);
    gemm_bf16_kernel<false><<<dim3(C / GBN, M / GBM), 128, 0, stream>>>(
        hbf, wkT, bk, nullptr, nullptr, kbf, M, C, C);
    gemm_bf16_kernel<false><<<dim3(C / GBN, M / GBM), 128, 0, stream>>>(
        hbf, wvT, bv, nullptr, nullptr, vbf, M, C, C);

    // attention (flash, online softmax); scale = 1/sqrt(64)
    attn_kernel<<<dim3(B * H, N / 128), 256, 0, stream>>>(
        qbf, kbf, vbf, obf, N, C, H, 0.125f);

    // output projection + residual(x) -> h2 (f32)
    gemm_bf16_kernel<false><<<dim3(C / GBN, M / GBM), 128, 0, stream>>>(
        obf, woT, bo, x, h2, nullptr, M, C, C);

    // ln2 -> bf16
    ln_kernel<<<M, 256, 0, stream>>>(h2, ln2w, ln2b, nullptr, mbf, C);

    // fc1 + exact GELU -> bf16
    gemm_bf16_kernel<true><<<dim3(C4 / GBN, M / GBM), 128, 0, stream>>>(
        mbf, fc1T, fc1b, nullptr, nullptr, gbf, M, C4, C);

    // fc2 + residual(h2) -> bf16 (only needed as next GEMM input)
    gemm_bf16_kernel<false><<<dim3(C / GBN, M / GBM), 128, 0, stream>>>(
        gbf, fc2T, fc2b, h2, nullptr, h3bf, M, C, C4);

    // wout projection -> f32
    gemm_bf16_kernel<false><<<dim3(C / GBN, M / GBM), 128, 0, stream>>>(
        h3bf, woutT, bout, nullptr, pf, nullptr, M, C, C);

    // final layernorm -> d_out (f32)
    ln_kernel<<<M, 256, 0, stream>>>(pf, lnow, lnob, (float*)d_out, nullptr, C);
}
